// SympleAgent_79826262164034
// MI455X (gfx1250) — matrix operand order repaired and checked by reading, not compiled, not run on hardware
//
#include <hip/hip_runtime.h>
#include <math.h>

typedef __bf16 bf16_t;
typedef __attribute__((ext_vector_type(16))) __bf16 v16bf;
typedef __attribute__((ext_vector_type(8)))  __bf16 v8bf;
typedef __attribute__((ext_vector_type(8)))  float  v8f;

#define DD    12
#define NODES 4095          // 2^12 - 1
#define BSZ   64
#define VV    128
#define HH    256
#define NOPS  20
#define KTOT  640           // 128 (x) + 2*256 (h_left, h_right)
#define NCOL  1280          // 768 iou + 256 f0 + 256 f1
#define NTIL  80            // NCOL / 16 column tiles
#define KCH   20            // KTOT / 32 K-chunks
#define ASTR  648           // padded bf16 LDS stride (1296B = 324 DW, 324%64=4 -> conflict-free)

// ---------------------------------------------------------------------------
// Weights stored FRAGMENT-TILED so each B-fragment load is one contiguous,
// fully coalesced 1KB-per-wave access:
//   Wtiled[((tile*KCH + kc)*32 + lane)*16 + e]
//     col = tile*16 + (lane&15)
//     k   = kc*32 + 16*(lane>>4) + e        (contiguous k order, same as A: lanes0-15 K 0..15,
//                                            lanes16-31 K 16..31, 2 vals/VGPR)
// Column groups: [0,768)=iou  [768,1024)=f0  [1024,1280)=f1
// K groups:      [0,128)=x    [128,384)=h_left  [384,640)=h_right
// ---------------------------------------------------------------------------
__global__ void prep_weights(const float* __restrict__ Wiou,
                             const float* __restrict__ Wf,
                             const float* __restrict__ Uiou,
                             const float* __restrict__ Uf,
                             bf16_t* __restrict__ Wtiled)
{
    int idx = blockIdx.x * blockDim.x + threadIdx.x;
    if (idx >= NCOL * KTOT) return;
    int e    = idx & 15;
    int t2   = idx >> 4;
    int lane = t2 & 31;
    int t3   = t2 >> 5;
    int kc   = t3 % KCH;
    int tile = t3 / KCH;

    int col = tile * 16 + (lane & 15);              // output column 0..1279
    int k   = kc * 32 + 16 * (lane >> 4) + e;       // reduction index 0..639

    float val;
    if (k < VV) {
        if (col < 3 * HH)        val = Wiou[k * (3 * HH) + col];
        else if (col < 4 * HH)   val = Wf[k * HH + (col - 3 * HH)];
        else                     val = Wf[k * HH + (col - 4 * HH)];
    } else {
        int kk = k - VV;
        int j  = kk >> 8;        // child index 0/1
        int hh = kk & (HH - 1);
        if (col < 3 * HH)        val = Uiou[(j * HH + hh) * (3 * HH) + col];
        else if (col < 4 * HH)   val = Uf[((0 * 2 + j) * HH + hh) * HH + (col - 3 * HH)];
        else                     val = Uf[((1 * 2 + j) * HH + hh) * HH + (col - 4 * HH)];
    }
    Wtiled[idx] = (bf16_t)val;
}

__global__ void prep_bias(const float* __restrict__ biou,
                          const float* __restrict__ bfv,
                          float* __restrict__ bAll)
{
    int n = blockIdx.x * blockDim.x + threadIdx.x;
    if (n >= NCOL) return;
    bAll[n] = (n < 3 * HH) ? biou[n] : bfv[(n - 3 * HH) & (HH - 1)];
}

// ---------------------------------------------------------------------------
// One tree level:  [rows,640]bf16 x [640,1280]bf16 -> fused LSTM gates.
// Workgroup = 8 wave32s, owns 16 rows x all 1280 columns.
// Wave w owns d-range [w*32, w*32+32): tiles (g,s) -> column g*256 + w*32 + s*16,
// so i/o/u/f0/f1 for a given (row,d) live in the SAME lane & VGPR slot across
// the 5 group accumulators -> lane-local gate fusion, no LDS round trip.
// All 10 B fragments are loaded into registers before the WMMA chain so the
// compiler can clause the loads and wait once per K-step.
// ---------------------------------------------------------------------------
__global__ __launch_bounds__(256)
void tree_level(const float*  __restrict__ argHot,
                const bf16_t* __restrict__ Wtiled,
                const float*  __restrict__ bAll,
                const bf16_t* __restrict__ hChild,
                const float*  __restrict__ cChild,
                bf16_t*       __restrict__ hOut,
                float*        __restrict__ cOut,
                int m, int lg, int Kmax, int leaf)
{
    __shared__ __align__(16) bf16_t Ap[16 * ASTR];   // 20,736 B

    const int tid = threadIdx.x;
    const int r0  = blockIdx.x * 16;

    // ---- Phase 1: stage A panel (x: f32->bf16 convert; h: gather children)
    for (int idx = tid; idx < 16 * Kmax; idx += 256) {
        int rr  = idx / Kmax;
        int k   = idx - rr * Kmax;
        int row = r0 + rr;
        int b   = row >> lg;
        int j   = row & (m - 1);
        bf16_t val;
        if (k < VV) {
            val = (bf16_t)argHot[((size_t)b * NODES + (m - 1) + j) * VV + k];
        } else if (!leaf) {
            int kk = k - VV;
            int ch = kk >> 8;
            int hh = kk & (HH - 1);
            val = hChild[((size_t)(b * 2 * m + 2 * j + ch)) * HH + hh];
        } else {
            val = (bf16_t)0.0f;
        }
        Ap[rr * ASTR + k] = val;
    }
    __syncthreads();

    const int lane  = tid & 31;
    const int wave  = tid >> 5;
    const int khalf = lane >> 4;      // A: K-half; C/D: M-half
    const int rowA  = lane & 15;
    const int colL  = lane & 15;

    // Fragment-tiled weight base for this wave/lane:
    //   tileIdx = g*16 + wave*2 + s ; element offset = tileIdx*KCH*512 + kc*512 + lane*16
    const bf16_t* wbase = Wtiled + (size_t)(wave * 2) * (KCH * 512) + lane * 16;

    v8f acc[10] = {};                 // acc[g*2+s], g in 0..4, s in 0..1

    // ---- Phase 2: K loop of bf16 WMMAs (Kmax=128 at leaves: children are 0)
    for (int kb = 0; kb < Kmax; kb += 32) {
        // A fragment, contiguous k order (same as B): VGPR0-3 = K kb+16*khalf+0..7,
        //                                  VGPR4-7 = K kb+16*khalf+8..15
        v8bf alo = *(const v8bf*)(Ap + rowA * ASTR + kb + 16 * khalf);
        v8bf ahi = *(const v8bf*)(Ap + rowA * ASTR + kb + 16 * khalf + 8);
        v16bf afrag = __builtin_shufflevector(alo, ahi,
            0, 1, 2, 3, 4, 5, 6, 7, 8, 9, 10, 11, 12, 13, 14, 15);

        const bf16_t* wk = wbase + (size_t)(kb >> 5) * 512;
        v16bf bfr[10];
        #pragma unroll
        for (int g = 0; g < 5; ++g) {
            #pragma unroll
            for (int s = 0; s < 2; ++s) {
                // (g*16 + s)*KCH*512 is a compile-time constant per (g,s):
                // folds into the 24-bit instruction offset.
                bfr[g * 2 + s] = *(const v16bf*)(wk + (size_t)(g * 16 + s) * (KCH * 512));
            }
        }
        #pragma unroll
        for (int t = 0; t < 10; ++t) {
            acc[t] = __builtin_amdgcn_wmma_f32_16x16x32_bf16(
                false, afrag, false, bfr[t],
                (short)0, acc[t], false, false);
        }
    }

    // ---- Phase 3: lane-local gate fusion + coalesced writeback
    #pragma unroll
    for (int s = 0; s < 2; ++s) {
        int d  = wave * 32 + s * 16 + colL;
        float bi = bAll[d];
        float bo = bAll[HH + d];
        float bu = bAll[2 * HH + d];
        float b0 = bAll[3 * HH + d];
        float b1 = bAll[4 * HH + d];
        #pragma unroll
        for (int r = 0; r < 8; ++r) {
            int rr  = r + 8 * khalf;          // C/D: M = r + 8*(lane>=16)
            int row = r0 + rr;
            int b   = row >> lg;
            int j   = row & (m - 1);
            float ig = acc[0 * 2 + s][r] + bi;
            float og = acc[1 * 2 + s][r] + bo;
            float ug = acc[2 * 2 + s][r] + bu;
            float f0 = acc[3 * 2 + s][r] + b0;
            float f1 = acc[4 * 2 + s][r] + b1;
            ig = 1.0f / (1.0f + __expf(-ig));
            og = 1.0f / (1.0f + __expf(-og));
            ug = tanhf(ug);
            f0 = 1.0f / (1.0f + __expf(-f0));
            f1 = 1.0f / (1.0f + __expf(-f1));
            float c0 = 0.0f, c1 = 0.0f;
            if (!leaf) {
                size_t cb = ((size_t)(b * 2 * m + 2 * j)) * HH + d;
                c0 = cChild[cb];
                c1 = cChild[cb + HH];
            }
            float c = ig * ug + f0 * c0 + f1 * c1;
            float h = og * tanhf(c);
            cOut[(size_t)row * HH + d] = c;
            hOut[(size_t)row * HH + d] = (bf16_t)h;
        }
    }
}

// ---------------------------------------------------------------------------
// Actor head: logits = log(vm) + (root_h . actor_w[o])*vm + actor_b[o]*vm,
// softmax(logits / 3). One wave32 per batch element.
// ---------------------------------------------------------------------------
__global__ void actor_softmax(const bf16_t* __restrict__ rootH,
                              const float*  __restrict__ actorW,
                              const float*  __restrict__ actorB,
                              const float*  __restrict__ vmask,
                              float*        __restrict__ out)
{
    int b    = blockIdx.x;
    int lane = threadIdx.x;
    float s  = -INFINITY;
    if (lane < NOPS) {
        float vm  = vmask[b * NOPS + lane];
        float dot = 0.0f;
        for (int h = 0; h < HH; ++h)
            dot += (float)rootH[b * HH + h] * actorW[lane * HH + h];
        float logit = __logf(vm) + dot * vm + actorB[lane] * vm;
        s = logit * (1.0f / 3.0f);
    }
    float mx = s;
    for (int off = 16; off > 0; off >>= 1)
        mx = fmaxf(mx, __shfl_xor(mx, off, 32));
    float e = (lane < NOPS) ? __expf(s - mx) : 0.0f;
    float sum = e;
    for (int off = 16; off > 0; off >>= 1)
        sum += __shfl_xor(sum, off, 32);
    if (lane < NOPS) out[b * NOPS + lane] = e / sum;
}

// ---------------------------------------------------------------------------
// Workspace layout (bytes, all 256-aligned):
//   Wtiled bf16: [0,            1,638,400)
//   bAll  f32  : [1,638,400,    1,643,520)
//   h1    bf16 : [1,643,520,   68,752,384)   (odd levels, sized for l=11)
//   h0    bf16 : [68,752,384, 102,306,816)   (even levels, sized for l=10)
//   c1    f32  : [102,306,816,236,524,544)
//   c0    f32  : [236,524,544,303,633,408)
// ---------------------------------------------------------------------------
extern "C" void kernel_launch(void* const* d_in, const int* in_sizes, int n_in,
                              void* d_out, int out_size, void* d_ws, size_t ws_size,
                              hipStream_t stream)
{
    const float* argHot = (const float*)d_in[0];
    const float* Wiou   = (const float*)d_in[1];
    const float* biou   = (const float*)d_in[2];
    const float* Uiou   = (const float*)d_in[3];
    const float* Wf     = (const float*)d_in[4];
    const float* bfv    = (const float*)d_in[5];
    const float* Uf     = (const float*)d_in[6];
    const float* actorW = (const float*)d_in[7];
    const float* actorB = (const float*)d_in[8];
    const float* vmask  = (const float*)d_in[9];
    float* out = (float*)d_out;

    char* ws = (char*)d_ws;
    bf16_t* Wtiled = (bf16_t*)(ws);
    float*  bAll   = (float*)(ws + 1638400);
    bf16_t* h1     = (bf16_t*)(ws + 1643520);
    bf16_t* h0     = (bf16_t*)(ws + 68752384);
    float*  c1     = (float*)(ws + 102306816);
    float*  c0     = (float*)(ws + 236524544);

    prep_weights<<<(NCOL * KTOT + 255) / 256, 256, 0, stream>>>(Wiou, Wf, Uiou, Uf, Wtiled);
    prep_bias<<<(NCOL + 255) / 256, 256, 0, stream>>>(biou, bfv, bAll);

    for (int l = DD - 1; l >= 0; --l) {
        int m    = 1 << l;
        int rows = BSZ * m;
        int leaf = (l == DD - 1);
        int Kmax = leaf ? VV : KTOT;          // children are exactly zero at leaves
        bf16_t* hC = (((l + 1) & 1) ? h1 : h0);
        float*  cC = (((l + 1) & 1) ? c1 : c0);
        bf16_t* hO = ((l & 1) ? h1 : h0);
        float*  cO = ((l & 1) ? c1 : c0);
        tree_level<<<rows / 16, 256, 0, stream>>>(argHot, Wtiled, bAll,
                                                  hC, cC, hO, cO,
                                                  m, l, Kmax, leaf);
    }
    // level 0 (root) h lives in h0
    actor_softmax<<<BSZ, 32, 0, stream>>>(h0, actorW, actorB, vmask, out);
}
